// MFPK_29910152249647
// MI455X (gfx1250) — matrix operand order repaired and checked by reading, not compiled, not run on hardware
//
#include <hip/hip_runtime.h>
#include <hip/hip_bf16.h>

// ---------------------------------------------------------------------------
// CDNA5 (gfx1250, wave32) implementation of the dual-branch molecular
// transformer. All matmuls use v_wmma_f32_16x16x32_bf16 (bf16 in, f32 acc).
// LDS tiles are stored in bf16 (converted once at staging time, using the
// native v_cvt_pk_bf16_f32 path) and laid out so every WMMA fragment is two
// contiguous 16-byte ds_load_b128s. Full-interior tiles take an unpredicated
// staging/epilogue fast path; edge tiles use the guarded path.
// ---------------------------------------------------------------------------

typedef __attribute__((ext_vector_type(16))) __bf16 v16bf;
typedef __attribute__((ext_vector_type(8)))  __bf16 v8bf;
typedef __attribute__((ext_vector_type(8)))  float  v8f;

struct bf16x16 { v8bf lo, hi; };

constexpr int NB  = 64;    // batch
constexpr int NS  = 128;   // sequence
constexpr int NF  = 62;    // input features
constexpr int ND  = 256;   // model dim
constexpr int NH  = 4;     // heads per branch
constexpr int NL  = 6;     // layers
constexpr int NSB = 32;    // NSUB
constexpr int NFF = 512;   // FFN dim
constexpr int NDH = ND / 2;      // 128 per branch
constexpr int NDEP = NDH / NH;   // 32 head depth
constexpr float NEGV = -1000000000.0f;
constexpr float SCALE = 0.17677669529663687f; // 1/sqrt(32)

__device__ inline __bf16 f2bf(float f) {
  __hip_bfloat16 h = __float2bfloat16(f);
  return __builtin_bit_cast(__bf16, h);
}

__device__ inline unsigned pack2(float a, float b) {
  unsigned short x = __builtin_bit_cast(unsigned short, f2bf(a));
  unsigned short y = __builtin_bit_cast(unsigned short, f2bf(b));
  return (unsigned)x | ((unsigned)y << 16);
}

// two contiguous 16-byte LDS runs -> one 16x32 (or 32x16) bf16 fragment
__device__ inline v16bf frag_from(const __bf16* p0, const __bf16* p1) {
  bf16x16 t;
  t.lo = *(const v8bf*)p0;
  t.hi = *(const v8bf*)p1;
  return __builtin_bit_cast(v16bf, t);
}

// ---------------------------------------------------------------------------
// Batched GEMM: C[bz] = act(A[bz] (MxK) * W[bz] (KxN) + bias)
// 128 threads = 4 waves; 64x64 tile; each wave a 32x32 quadrant (2x2 frags);
// K stepped by 32 through bf16 LDS. act: 0=none, 1=relu, 2=gelu(exact)
// ---------------------------------------------------------------------------
__global__ __launch_bounds__(128) void gemm_bf16_wmma(
    const float* __restrict__ A, int lda, long long sA,
    const float* __restrict__ Bm, int ldb, long long sB,
    const float* __restrict__ bias,
    float* __restrict__ C, int ldc, long long sC,
    int M, int N, int K, int act)
{
  __shared__ __align__(16) __bf16 As[64][40];   // row-major, k contiguous
  __shared__ __align__(16) __bf16 Bst[64][40];  // transposed: [n][k], k contiguous

  const int bz = blockIdx.z;
  A  += (size_t)bz * sA;
  Bm += (size_t)bz * sB;
  C  += (size_t)bz * sC;

  const int n0 = blockIdx.x * 64;
  const int m0 = blockIdx.y * 64;
  const int tid  = threadIdx.x;
  const int lane = tid & 31;
  const int w    = tid >> 5;
  const int half = lane >> 4;      // 0: lanes 0-15, 1: lanes 16-31
  const int ln   = lane & 15;
  const int wr = (w >> 1) * 32;    // wave row offset in tile
  const int wc = (w & 1) * 32;     // wave col offset in tile
  const bool interior = (m0 + 64 <= M) && (n0 + 64 <= N);

  const v8f zero8 = {0.f,0.f,0.f,0.f,0.f,0.f,0.f,0.f};
  v8f acc[2][2];
  acc[0][0] = zero8; acc[0][1] = zero8; acc[1][0] = zero8; acc[1][1] = zero8;

  for (int kk = 0; kk < K; kk += 32) {
    if (interior && (kk + 32 <= K)) {
      // ---- fast path: unpredicated, batched loads ----
      // A: thread -> row r, 16-col segment; 8 x float2 (8B-aligned: all offsets even)
      {
        const int r = tid >> 1, cseg = (tid & 1) * 16;
        const float* ap = &A[(size_t)(m0 + r) * lda + kk + cseg];
        float2 av[8];
#pragma unroll
        for (int i = 0; i < 8; ++i) av[i] = ((const float2*)ap)[i];
#pragma unroll
        for (int i = 0; i < 8; ++i)
          *(unsigned*)&As[r][cseg + 2 * i] = pack2(av[i].x, av[i].y);
      }
      // B: thread -> column n, 16-row (k) segment; strided loads, coalesced per k
      {
        const int n = tid >> 1, kseg = (tid & 1) * 16;
        const float* bp = &Bm[(size_t)(kk + kseg) * ldb + n0 + n];
        float bv[16];
#pragma unroll
        for (int i = 0; i < 16; ++i) bv[i] = bp[(size_t)i * ldb];
#pragma unroll
        for (int i = 0; i < 8; ++i)
          *(unsigned*)&Bst[n][kseg + 2 * i] = pack2(bv[2 * i], bv[2 * i + 1]);
      }
    } else {
      // ---- edge path: guarded per element ----
      for (int idx = tid; idx < 64 * 16; idx += 128) {
        int r = idx >> 4, cp = (idx & 15) * 2;
        int gm = m0 + r, gk = kk + cp;
        float a0 = (gm < M && gk     < K) ? A[(size_t)gm * lda + gk]     : 0.f;
        float a1 = (gm < M && gk + 1 < K) ? A[(size_t)gm * lda + gk + 1] : 0.f;
        *(unsigned*)&As[r][cp] = pack2(a0, a1);
      }
      for (int idx = tid; idx < 64 * 16; idx += 128) {
        int n = idx >> 4, kp = (idx & 15) * 2;
        int gn = n0 + n, gk = kk + kp;
        float b0 = (gk     < K && gn < N) ? Bm[(size_t)gk * ldb + gn]       : 0.f;
        float b1 = (gk + 1 < K && gn < N) ? Bm[(size_t)(gk + 1) * ldb + gn] : 0.f;
        *(unsigned*)&Bst[n][kp] = pack2(b0, b1);
      }
    }
    // prefetch next K tile into cache
    if (kk + 32 < K) {
      int gm = m0 + (tid >> 1);
      if (gm < M) __builtin_prefetch(&A[(size_t)gm * lda + kk + 32], 0, 1);
      int gk = kk + 32 + (tid >> 2);
      if (gk < K) __builtin_prefetch(&Bm[(size_t)gk * ldb + n0], 0, 1);
    }
    __syncthreads();

    v16bf af[2], bfm[2];
#pragma unroll
    for (int f = 0; f < 2; ++f) {
      const int m = wr + f * 16 + ln;
      af[f] = frag_from(&As[m][half * 8], &As[m][16 + half * 8]);
      const int n = wc + f * 16 + ln;
      bfm[f] = frag_from(&Bst[n][half * 8], &Bst[n][16 + half * 8]);
    }
#pragma unroll
    for (int i = 0; i < 2; ++i)
#pragma unroll
      for (int j = 0; j < 2; ++j)
        acc[i][j] = __builtin_amdgcn_wmma_f32_16x16x32_bf16(
            false, af[i], false, bfm[j], (short)0, acc[i][j], false, false);
    __syncthreads();
  }

  // epilogue: bias + activation, C-frag layout -> global
#pragma unroll
  for (int j = 0; j < 2; ++j) {
    const int gnw = wc + j * 16 + ln;              // column within tile
    const int gn  = n0 + gnw;
    float bb = 0.f;
    if (bias && (interior || gn < N)) bb = bias[gn];
#pragma unroll
    for (int i = 0; i < 2; ++i)
#pragma unroll
      for (int e = 0; e < 8; ++e) {
        const int gm = m0 + wr + i * 16 + half * 8 + e;
        float v = acc[i][j][e] + bb;
        if (act == 1)      v = fmaxf(v, 0.f);
        else if (act == 2) v = 0.5f * v * (1.f + erff(v * 0.70710678118654752f));
        if (interior || (gm < M && gn < N))
          C[(size_t)gm * ldc + gn] = v;
      }
  }
}

// ---------------------------------------------------------------------------
// Fused dual-branch attention. grid = (NB*NH, 2 branches), 128 threads.
// Each wave owns 32 query rows; full 128x128 score matrix kept in C-frags.
// Q/K staged row-major bf16 (k contiguous); V staged transposed Vst[d][j];
// softmaxed P written back to LDS in bf16 for the P*V pass.
// ---------------------------------------------------------------------------
__global__ __launch_bounds__(128) void attn_kernel(
    const float* __restrict__ Q1, const float* __restrict__ K1, const float* __restrict__ V1,
    const float* __restrict__ Q2, const float* __restrict__ K2, const float* __restrict__ V2,
    const float* __restrict__ adj, const float* __restrict__ dist,
    const float* __restrict__ maskv, float* __restrict__ O)
{
  __shared__ __align__(16) __bf16 Qs[128][40];   // [s][d]
  __shared__ __align__(16) __bf16 Ks[128][40];   // [s][d]
  __shared__ __align__(16) __bf16 Vst[32][136];  // [d][j]

  const int bh = blockIdx.x;
  const int b  = bh / NH;
  const int hh = bh % NH;
  const int branch = blockIdx.y;

  const float* Q  = branch ? Q2 : Q1;
  const float* Kd = branch ? K2 : K1;
  const float* V  = branch ? V2 : V1;

  const int tid  = threadIdx.x;
  const int lane = tid & 31;
  const int w    = tid >> 5;
  const int half = lane >> 4;
  const int ln   = lane & 15;

  // stage the (128 x 32) head slices of Q, K, V (bf16)
  for (int idx = tid; idx < NS * NDEP / 2; idx += 128) {
    int s = idx >> 4, dp = (idx & 15) * 2;
    size_t g = ((size_t)(b * NS + s)) * NDH + hh * NDEP + dp;
    float q0 = Q[g],  q1 = Q[g + 1];
    float k0 = Kd[g], k1 = Kd[g + 1];
    float v0 = V[g],  v1 = V[g + 1];
    *(unsigned*)&Qs[s][dp] = pack2(q0, q1);
    *(unsigned*)&Ks[s][dp] = pack2(k0, k1);
    Vst[dp][s]     = f2bf(v0);
    Vst[dp + 1][s] = f2bf(v1);
  }
  __syncthreads();

  // Q A-frags for this wave's 32 rows (K = DEPTH = 32, single step)
  v16bf aq[2];
#pragma unroll
  for (int rf = 0; rf < 2; ++rf) {
    const int m = w * 32 + rf * 16 + ln;
    aq[rf] = frag_from(&Qs[m][half * 8], &Qs[m][16 + half * 8]);
  }

  const v8f zero8 = {0.f,0.f,0.f,0.f,0.f,0.f,0.f,0.f};
  v8f sc[2][8];

  // scores: S = Q (32x32) * K^T (32x128) for this wave
#pragma unroll
  for (int cf = 0; cf < 8; ++cf) {
    const int j = cf * 16 + ln;
    v16bf bk = frag_from(&Ks[j][half * 8], &Ks[j][16 + half * 8]); // B(k,n)=K[j][k]
#pragma unroll
    for (int rf = 0; rf < 2; ++rf)
      sc[rf][cf] = __builtin_amdgcn_wmma_f32_16x16x32_bf16(
          false, aq[rf], false, bk, (short)0, zero8, false, false);
  }

  // per-column mask bias (depends on j only) hoisted out of the element loops
  float mkv[8];
#pragma unroll
  for (int cf = 0; cf < 8; ++cf) mkv[cf] = maskv[b * NS + cf * 16 + ln] * NEGV;

  // branch epilogue + row softmax (row lives in 16 lanes of one half-group)
#pragma unroll
  for (int rf = 0; rf < 2; ++rf) {
#pragma unroll
    for (int e = 0; e < 8; ++e) {
      const int i = w * 32 + rf * 16 + half * 8 + e;
      const float* arow = &adj[((size_t)(b * NS + i)) * NS];
      const float* drow = &dist[((size_t)(b * NS + i)) * NS];
      float rowv[8];
#pragma unroll
      for (int cf = 0; cf < 8; ++cf) {
        const int j = cf * 16 + ln;
        float v = sc[rf][cf][e];
        if (branch == 0) {
          v = v * SCALE + mkv[cf] + arow[j];
        } else {
          v = fmaxf(v, 0.f) * drow[j] * SCALE + mkv[cf];
        }
        rowv[cf] = v;
      }
      float mx = rowv[0];
#pragma unroll
      for (int cf = 1; cf < 8; ++cf) mx = fmaxf(mx, rowv[cf]);
#pragma unroll
      for (int off = 8; off >= 1; off >>= 1) mx = fmaxf(mx, __shfl_xor(mx, off, 16));
      float sum = 0.f;
#pragma unroll
      for (int cf = 0; cf < 8; ++cf) { rowv[cf] = __expf(rowv[cf] - mx); sum += rowv[cf]; }
#pragma unroll
      for (int off = 8; off >= 1; off >>= 1) sum += __shfl_xor(sum, off, 16);
      const float inv = 1.f / sum;
#pragma unroll
      for (int cf = 0; cf < 8; ++cf) sc[rf][cf][e] = rowv[cf] * inv;
    }
  }

  // O = P (32x128) * V (128x32): P goes through LDS already in bf16 (reuse Qs)
  __bf16 (*Ps)[40] = Qs;
  v8f oacc[2][2];
  oacc[0][0] = zero8; oacc[0][1] = zero8; oacc[1][0] = zero8; oacc[1][1] = zero8;
  __syncthreads();

  for (int c = 0; c < 4; ++c) {
    // store 32x32 P chunk (key cols c*32..c*32+31) to this wave's LDS slab
#pragma unroll
    for (int rf = 0; rf < 2; ++rf)
#pragma unroll
      for (int cc = 0; cc < 2; ++cc) {
        const int cf = c * 2 + cc;
#pragma unroll
        for (int e = 0; e < 8; ++e) {
          const int pm = rf * 16 + half * 8 + e;
          const int pn = cc * 16 + ln;
          Ps[w * 32 + pm][pn] = f2bf(sc[rf][cf][e]);
        }
      }
    __syncthreads();
    v16bf bv[2];
#pragma unroll
    for (int cf2 = 0; cf2 < 2; ++cf2) {
      const int n = cf2 * 16 + ln;   // output depth dim
      bv[cf2] = frag_from(&Vst[n][c * 32 + half * 8], &Vst[n][c * 32 + 16 + half * 8]);
    }
#pragma unroll
    for (int rf = 0; rf < 2; ++rf) {
      const int m = w * 32 + rf * 16 + ln;
      v16bf ap = frag_from(&Ps[m][half * 8], &Ps[m][16 + half * 8]);
#pragma unroll
      for (int cf2 = 0; cf2 < 2; ++cf2)
        oacc[rf][cf2] = __builtin_amdgcn_wmma_f32_16x16x32_bf16(
            false, ap, false, bv[cf2], (short)0, oacc[rf][cf2], false, false);
    }
    __syncthreads();
  }

  // write O into (B,S,D) with branch column offset
#pragma unroll
  for (int rf = 0; rf < 2; ++rf)
#pragma unroll
    for (int cf2 = 0; cf2 < 2; ++cf2)
#pragma unroll
      for (int e = 0; e < 8; ++e) {
        const int i = w * 32 + rf * 16 + half * 8 + e;
        const int d = cf2 * 16 + ln;
        O[((size_t)(b * NS + i)) * ND + branch * NDH + hh * NDEP + d] = oacc[rf][cf2][e];
      }
}

// ---------------------------------------------------------------------------
// out = LayerNorm(a + r) * g + be     (one 256-wide row per 256-thread block)
// ---------------------------------------------------------------------------
__global__ __launch_bounds__(256) void add_ln_kernel(
    const float* __restrict__ a, const float* __restrict__ r,
    const float* __restrict__ g, const float* __restrict__ be,
    float* __restrict__ out, int d)
{
  const int row = blockIdx.x;
  const int tid = threadIdx.x;
  const float v = a[(size_t)row * d + tid] + r[(size_t)row * d + tid];

  __shared__ float s1[8], s2[8];
  float sum = v, sq = v * v;
#pragma unroll
  for (int off = 16; off >= 1; off >>= 1) {
    sum += __shfl_xor(sum, off, 32);
    sq  += __shfl_xor(sq,  off, 32);
  }
  const int lane = tid & 31, w = tid >> 5;
  if (lane == 0) { s1[w] = sum; s2[w] = sq; }
  __syncthreads();
  if (w == 0) {
    float t1 = (lane < 8) ? s1[lane] : 0.f;
    float t2 = (lane < 8) ? s2[lane] : 0.f;
#pragma unroll
    for (int off = 4; off >= 1; off >>= 1) { t1 += __shfl_xor(t1, off, 32); t2 += __shfl_xor(t2, off, 32); }
    if (lane == 0) { s1[0] = t1; s2[0] = t2; }
  }
  __syncthreads();
  const float mean = s1[0] / d;
  const float var  = s2[0] / d - mean * mean;
  const float inv  = rsqrtf(var + 1e-6f);
  out[(size_t)row * d + tid] = (v - mean) * inv * g[tid] + be[tid];
}

// mask[b*S+s] = (sum_f x == 0) ? 1 : 0
__global__ void mask_kernel(const float* __restrict__ x, float* __restrict__ m, int rows) {
  int i = blockIdx.x * blockDim.x + threadIdx.x;
  if (i < rows) {
    float s = 0.f;
    for (int f = 0; f < NF; ++f) s += x[(size_t)i * NF + f];
    m[i] = (s == 0.f) ? 1.f : 0.f;
  }
}

// dist_rescaled = (1 + e) / (1 + exp(1 - w))
__global__ void rescale_kernel(const float* __restrict__ w, float* __restrict__ o, long long n) {
  long long i = (long long)blockIdx.x * blockDim.x + threadIdx.x;
  if (i < n) {
    const float e1 = 2.7182818284590452f;
    o[i] = (1.f + e1) / (1.f + __expf(1.f - w[i]));
  }
}

// pooled[b][m][:] /= sum_atoms[b][m]
__global__ void rowdiv_kernel(float* __restrict__ p, const float* __restrict__ s, long long n, int d) {
  long long i = (long long)blockIdx.x * blockDim.x + threadIdx.x;
  if (i < n) p[i] /= s[i / d];
}

// ---------------------------------------------------------------------------
extern "C" void kernel_launch(void* const* d_in, const int* in_sizes, int n_in,
                              void* d_out, int out_size, void* d_ws, size_t ws_size,
                              hipStream_t stream)
{
  (void)in_sizes; (void)n_in; (void)out_size; (void)ws_size;

  const float* x        = (const float*)d_in[0];
  const float* adj      = (const float*)d_in[1];
  const float* dist_raw = (const float*)d_in[2];
  const float* amm      = (const float*)d_in[3];
  const float* sum_at   = (const float*)d_in[4];
  const float* emb_W    = (const float*)d_in[5];
  const float* emb_b    = (const float*)d_in[6];
  const float* glob_W   = (const float*)d_in[7];
  const float* glob_b   = (const float*)d_in[8];
  const float* Wq1 = (const float*)d_in[9];
  const float* Wk1 = (const float*)d_in[10];
  const float* Wv1 = (const float*)d_in[11];
  const float* Wo1 = (const float*)d_in[12];
  const float* Wq2 = (const float*)d_in[13];
  const float* Wk2 = (const float*)d_in[14];
  const float* Wv2 = (const float*)d_in[15];
  const float* Wo2 = (const float*)d_in[16];
  const float* bq1 = (const float*)d_in[17];
  const float* bk1 = (const float*)d_in[18];
  const float* bv1 = (const float*)d_in[19];
  const float* bo1 = (const float*)d_in[20];
  const float* bq2 = (const float*)d_in[21];
  const float* bk2 = (const float*)d_in[22];
  const float* bv2 = (const float*)d_in[23];
  const float* bo2 = (const float*)d_in[24];
  const float* Wf1 = (const float*)d_in[25];
  const float* bf1 = (const float*)d_in[26];
  const float* Wf2 = (const float*)d_in[27];
  const float* bf2 = (const float*)d_in[28];
  const float* g1  = (const float*)d_in[29];
  const float* be1 = (const float*)d_in[30];
  const float* g2  = (const float*)d_in[31];
  const float* be2 = (const float*)d_in[32];

  float* ws = (float*)d_ws;
  size_t off = 0;
  auto alloc = [&](size_t n) { float* p = ws + off; off += n; return p; };

  float* maskb = alloc((size_t)NB * NS);
  float* dist  = alloc((size_t)NB * NS * NS);
  float* h     = alloc((size_t)NB * NS * ND);
  float* Q1b   = alloc((size_t)NB * NS * NDH);
  float* K1b   = alloc((size_t)NB * NS * NDH);
  float* V1b   = alloc((size_t)NB * NS * NDH);
  float* Q2b   = alloc((size_t)NB * NS * NDH);
  float* K2b   = alloc((size_t)NB * NS * NDH);
  float* V2b   = alloc((size_t)NB * NS * NDH);
  float* Obuf  = alloc((size_t)NB * NS * ND);
  float* attnb = alloc((size_t)NB * NS * ND);
  float* out1  = alloc((size_t)NB * NS * ND);
  float* midb   = Q1b;   // FFN mid (B*S*NFF = 4.2M floats) aliases QKV region (6.3M floats)
  float* ffnb   = attnb; // attn consumed by first LN before FFN output is written
  float* pooled = Obuf;  // O buffer dead after final layer

  auto gemm = [&](const float* A, int lda, long long sA,
                  const float* Bm, int ldb, long long sB,
                  const float* bias, float* C, int ldc, long long sC,
                  int M, int N, int K, int batch, int act) {
    dim3 g((N + 63) / 64, (M + 63) / 64, batch);
    gemm_bf16_wmma<<<g, 128, 0, stream>>>(A, lda, sA, Bm, ldb, sB, bias, C, ldc, sC, M, N, K, act);
  };

  // prologue
  mask_kernel<<<(NB * NS + 255) / 256, 256, 0, stream>>>(x, maskb, NB * NS);
  {
    long long n = (long long)NB * NS * NS;
    rescale_kernel<<<(unsigned)((n + 255) / 256), 256, 0, stream>>>(dist_raw, dist, n);
  }
  // h = relu(x @ emb_W + emb_b)
  gemm(x, NF, 0, emb_W, ND, 0, emb_b, h, ND, 0, NB * NS, ND, NF, 1, 1);

  for (int l = 0; l < NL; ++l) {
    const size_t wO = (size_t)l * NDH * NDH;
    const size_t bO = (size_t)l * NDH;
    // QKV projections (x1 = h[:, :128], x2 = h[:, 128:], lda = ND)
    gemm(h,        ND, 0, Wq1 + wO, NDH, 0, bq1 + bO, Q1b, NDH, 0, NB * NS, NDH, NDH, 1, 0);
    gemm(h,        ND, 0, Wk1 + wO, NDH, 0, bk1 + bO, K1b, NDH, 0, NB * NS, NDH, NDH, 1, 0);
    gemm(h,        ND, 0, Wv1 + wO, NDH, 0, bv1 + bO, V1b, NDH, 0, NB * NS, NDH, NDH, 1, 0);
    gemm(h + NDH,  ND, 0, Wq2 + wO, NDH, 0, bq2 + bO, Q2b, NDH, 0, NB * NS, NDH, NDH, 1, 0);
    gemm(h + NDH,  ND, 0, Wk2 + wO, NDH, 0, bk2 + bO, K2b, NDH, 0, NB * NS, NDH, NDH, 1, 0);
    gemm(h + NDH,  ND, 0, Wv2 + wO, NDH, 0, bv2 + bO, V2b, NDH, 0, NB * NS, NDH, NDH, 1, 0);

    // fused dual-branch attention
    attn_kernel<<<dim3(NB * NH, 2), 128, 0, stream>>>(
        Q1b, K1b, V1b, Q2b, K2b, V2b, adj, dist, maskb, Obuf);

    // output projections into concat(attn)
    gemm(Obuf,       ND, 0, Wo1 + wO, NDH, 0, bo1 + bO, attnb,       ND, 0, NB * NS, NDH, NDH, 1, 0);
    gemm(Obuf + NDH, ND, 0, Wo2 + wO, NDH, 0, bo2 + bO, attnb + NDH, ND, 0, NB * NS, NDH, NDH, 1, 0);

    // out1 = LN(h + attn)
    add_ln_kernel<<<NB * NS, 256, 0, stream>>>(h, attnb, g1 + (size_t)l * ND, be1 + (size_t)l * ND, out1, ND);

    // FFN: mid = gelu(out1 @ Wf1 + bf1); ffn = mid @ Wf2 + bf2
    gemm(out1, ND, 0, Wf1 + (size_t)l * ND * NFF, NFF, 0, bf1 + (size_t)l * NFF,
         midb, NFF, 0, NB * NS, NFF, ND, 1, 2);
    gemm(midb, NFF, 0, Wf2 + (size_t)l * NFF * ND, ND, 0, bf2 + (size_t)l * ND,
         ffnb, ND, 0, NB * NS, ND, NFF, 1, 0);

    // h = LN(ffn + out1)
    add_ln_kernel<<<NB * NS, 256, 0, stream>>>(ffnb, out1, g2 + (size_t)l * ND, be2 + (size_t)l * ND, h, ND);
  }

  // pooled[b] = atom_match[b] (32x128) @ h[b] (128x256)
  gemm(amm, NS, (long long)NSB * NS, h, ND, (long long)NS * ND, nullptr,
       pooled, ND, (long long)NSB * ND, NSB, ND, NS, NB, 0);
  {
    long long n = (long long)NB * NSB * ND;
    rowdiv_kernel<<<(unsigned)((n + 255) / 256), 256, 0, stream>>>(pooled, sum_at, n, ND);
  }
  // out = relu(pooled @ glob_W + glob_b)
  gemm(pooled, ND, 0, glob_W, ND, 0, glob_b, (float*)d_out, ND, 0, NB * NSB, ND, ND, 1, 1);
}
